// NetMHCpan_10_LSTM_584115552896
// MI455X (gfx1250) — compile-verified, matrix operand order checked by reading, not run on hardware
//
#include <hip/hip_runtime.h>
#include <hip/hip_bf16.h>
#include <math.h>

// ---------------------------------------------------------------------------
// NetMHCpan LSTM+MLP on gfx1250 (MI455X, wave32)
//  - v_wmma_f32_16x16x32_f16 matrix core path
//  - Tensor Data Mover (tensor_load_to_lds) double-buffered LDS staging
//  - 128x16 per-block recurrent tile, 8 WMMA accumulators per wave
// ---------------------------------------------------------------------------

typedef _Float16 v16h __attribute__((ext_vector_type(16)));
typedef _Float16 v8h  __attribute__((ext_vector_type(8)));
typedef float    v8f  __attribute__((ext_vector_type(8)));
typedef unsigned v4u  __attribute__((ext_vector_type(4)));
typedef unsigned v8u  __attribute__((ext_vector_type(8)));

#define B_SZ   4096
#define H_LSTM 512
#define G4     2048   // 4*H_LSTM
#define H_MLP  1024
#define L_PEP  15
#define L_HLA  34
#define KPAD   32     // input dim 20 padded to 32
#define MT     128    // M tile per block (rows of batch)

// LDS tile row stride: 32 f16 (16 DW) + TDM pad of 4 DW = 20 DW = 40 halves = 80B
#define ROWH   40

// --- WMMA fragment helpers (ISA 7.12.2 layout) -----------------------------
__device__ __forceinline__ v16h load_frag_g(const _Float16* __restrict__ base,
                                            int stride, int row, int k0, int lane) {
    const int kh = (lane >> 4) << 3;
    const _Float16* p = base + (long)row * stride + k0 + kh;
    v8h lo = *(const v8h*)(p);
    v8h hi = *(const v8h*)(p + 16);
    return __builtin_shufflevector(lo, hi, 0,1,2,3,4,5,6,7,8,9,10,11,12,13,14,15);
}
__device__ __forceinline__ v16h load_frag_s(const _Float16* base, int row, int lane) {
    const int kh = (lane >> 4) << 3;
    const _Float16* p = base + row * ROWH + kh;
    v8h lo = *(const v8h*)(p);
    v8h hi = *(const v8h*)(p + 16);
    return __builtin_shufflevector(lo, hi, 0,1,2,3,4,5,6,7,8,9,10,11,12,13,14,15);
}

__device__ __forceinline__ v8f wmma_f16(v16h a, v16h b, v8f c) {
    return __builtin_amdgcn_wmma_f32_16x16x32_f16(false, a, false, b,
                                                  (short)0, c, false, false);
}

__device__ __forceinline__ float sigmoidf_fast(float x) {
    return 1.0f / (1.0f + __expf(-x));
}

__device__ __forceinline__ unsigned lds_off_of(const void* p) {
    return (unsigned)(unsigned long long)
        (const __attribute__((address_space(3))) void*)p;
}

// --- Tensor Data Mover: 2-D f16 tile load, global -> LDS -------------------
// D# per CDNA5 ISA, 2-group form (<=2D tensor).
// pad: 4 DWORDs after every 16 DWORDs -> LDS row stride 80B (= ROWH halves).
__device__ __forceinline__ void tdm_load_2d_f16(
    unsigned lds_byte_off, const _Float16* gptr,
    unsigned tile_k, unsigned tile_rows,
    unsigned row_stride_elems, unsigned tdim0, unsigned tdim1)
{
    unsigned long long ga = (unsigned long long)gptr;
    v4u g0;
    g0[0] = 1u;                                  // count=1, user mode
    g0[1] = lds_byte_off;                        // lds_addr
    g0[2] = (unsigned)(ga & 0xffffffffu);        // global_addr[31:0]
    g0[3] = (unsigned)((ga >> 32) & 0x01ffffffu) // global_addr[56:32]
          | (2u << 30);                          // type = 2 ("image")
    v8u g1;
    g1[0] = (1u << 16)                           // data_size = 1 (2 bytes)
          | (1u << 20)                           // pad_enable
          | (3u << 22)                           // pad_interval: 16 DWORDs
          | (3u << 25);                          // pad_amount: 4 DWORDs
    g1[1] = (tdim0 & 0xffffu) << 16;             // tensor_dim0[15:0]
    g1[2] = (tdim0 >> 16) | ((tdim1 & 0xffffu) << 16);
    g1[3] = (tdim1 >> 16) | (tile_k << 16);      // tile_dim0
    g1[4] = tile_rows & 0xffffu;                 // tile_dim1 (tile_dim2=0)
    g1[5] = row_stride_elems;                    // tensor_dim0_stride[31:0]
    g1[6] = 0u;
    g1[7] = 0u;
    asm volatile("tensor_load_to_lds %0, %1" :: "s"(g0), "s"(g1) : "memory");
}

// --- prep kernels ----------------------------------------------------------
__global__ void k_cvt_f16(const float* __restrict__ s, _Float16* __restrict__ d, int n) {
    for (int i = blockIdx.x * blockDim.x + threadIdx.x; i < n; i += gridDim.x * blockDim.x)
        d[i] = (_Float16)s[i];
}
__global__ void k_cvt_pad20(const float* __restrict__ s, _Float16* __restrict__ d, int rows) {
    int n = rows * KPAD;
    for (int i = blockIdx.x * blockDim.x + threadIdx.x; i < n; i += gridDim.x * blockDim.x) {
        int col = i & (KPAD - 1), row = i >> 5;
        d[i] = (col < 20) ? (_Float16)s[row * 20 + col] : (_Float16)0.0f;
    }
}
__global__ void k_bias_comb(const float* __restrict__ a, const float* __restrict__ b,
                            float* __restrict__ o, int n) {
    for (int i = blockIdx.x * blockDim.x + threadIdx.x; i < n; i += gridDim.x * blockDim.x)
        o[i] = a[i] + b[i];
}
__global__ void k_zero32(unsigned int* __restrict__ p, int n) {
    for (int i = blockIdx.x * blockDim.x + threadIdx.x; i < n; i += gridDim.x * blockDim.x)
        p[i] = 0u;
}

// --- one LSTM recurrence step ----------------------------------------------
// grid (4096/128, 512/16), block 128 (4 waves). Wave w owns gate w's 128x16
// slab (8 WMMA accumulators). TDM double-buffers the shared 128x32 A tile
// (h_prev, issued by wave 0) and each wave's 16x32 B tile (W_hh rows);
// chunk i+1 DMA overlaps the 8 WMMAs of chunk i.
__global__ __launch_bounds__(128) void k_lstm_step(
    const _Float16* __restrict__ Hprev,  // [4096][512] f16
    const _Float16* __restrict__ Whh,    // [2048][512] f16
    const _Float16* __restrict__ Xt,     // [4096][32]  f16 (padded)
    const _Float16* __restrict__ Wih,    // [2048][32]  f16 (padded)
    const float*    __restrict__ bias,   // [2048] (b_ih+b_hh)
    float*          __restrict__ C,      // [4096][512] f32, in place
    _Float16*       __restrict__ Hnew)   // [4096][512] f16
{
    __shared__ _Float16 shA[2][MT * ROWH + 8];      // padded A tiles (10KB ea)
    __shared__ _Float16 shB[2][4][16 * ROWH + 8];   // padded B tiles per wave
    __shared__ float    shG[4 * MT * 16];           // gate exchange (32KB)

    const int tid  = threadIdx.x;
    const int lane = tid & 31;
    const int col  = lane & 15;
    const int m0   = blockIdx.x * MT;
    const int n0   = blockIdx.y * 16;
    // wave id as a *provably uniform* (SGPR) value for TDM descriptors
    const int wgate = __builtin_amdgcn_readfirstlane(tid >> 5);
    const int wrow  = wgate * H_LSTM + n0 + col;
    const _Float16* wbase = Whh + (size_t)(wgate * H_LSTM + n0) * H_LSTM;
    const _Float16* abase = Hprev + (size_t)m0 * H_LSTM;

    const unsigned offA0 = lds_off_of(&shA[0][0]);
    const unsigned offA1 = lds_off_of(&shA[1][0]);
    const unsigned offB0 = lds_off_of(&shB[0][wgate][0]);
    const unsigned offB1 = lds_off_of(&shB[1][wgate][0]);

    // --- TDM: kick K-chunk 0 immediately; overlap with input projection ---
    if (wgate == 0)
        tdm_load_2d_f16(offA0, abase, 32, MT, H_LSTM, H_LSTM, B_SZ);
    tdm_load_2d_f16(offB0, wbase, 32, 16, H_LSTM, H_LSTM, G4);

    v8f acc[8] = {};

    // input projection: K=32 (k<20 nonzero), straight from global
    {
        v16h bfrag = load_frag_g(Wih, KPAD, wrow, 0, lane);
        #pragma unroll
        for (int a = 0; a < 8; ++a) {
            v16h afrag = load_frag_g(Xt, KPAD, m0 + a * 16 + col, 0, lane);
            acc[a] = wmma_f16(afrag, bfrag, acc[a]);
        }
    }

    // --- recurrent GEMM: K = 512, 16 chunks of 32, 2-phase pipeline ---
    #pragma unroll 1
    for (int i = 0; i < 16; i += 2) {
        // ---- even chunk i: data in buffer 0 ----
        __builtin_amdgcn_s_wait_tensorcnt(0);
        __syncthreads();
        {   // prefetch chunk i+1 (always exists: i <= 14)
            const int k1 = (i + 1) * 32;
            if (wgate == 0)
                tdm_load_2d_f16(offA1, abase + k1, 32, MT, H_LSTM, H_LSTM, B_SZ);
            tdm_load_2d_f16(offB1, wbase + k1, 32, 16, H_LSTM, H_LSTM, G4);
        }
        {
            v16h bfrag = load_frag_s(&shB[0][wgate][0], col, lane);
            #pragma unroll
            for (int a = 0; a < 8; ++a) {
                v16h afrag = load_frag_s(&shA[0][0], a * 16 + col, lane);
                acc[a] = wmma_f16(afrag, bfrag, acc[a]);
            }
        }

        // ---- odd chunk i+1: data in buffer 1 ----
        __builtin_amdgcn_s_wait_tensorcnt(0);
        __syncthreads();
        if (i + 2 < 16) {   // prefetch chunk i+2
            const int k2 = (i + 2) * 32;
            if (wgate == 0)
                tdm_load_2d_f16(offA0, abase + k2, 32, MT, H_LSTM, H_LSTM, B_SZ);
            tdm_load_2d_f16(offB0, wbase + k2, 32, 16, H_LSTM, H_LSTM, G4);
        }
        {
            v16h bfrag = load_frag_s(&shB[1][wgate][0], col, lane);
            #pragma unroll
            for (int a = 0; a < 8; ++a) {
                v16h afrag = load_frag_s(&shA[1][0], a * 16 + col, lane);
                acc[a] = wmma_f16(afrag, bfrag, acc[a]);
            }
        }
    }

    // bias + stash this gate's 128x16 slab in LDS
    const float bv = bias[wrow];
    #pragma unroll
    for (int a = 0; a < 8; ++a) {
        #pragma unroll
        for (int r = 0; r < 8; ++r) {
            int row = a * 16 + r + ((lane >> 4) << 3);  // C/D: M = r + 8*(lane>>4)
            shG[wgate * (MT * 16) + row * 16 + col] = acc[a][r] + bv;
        }
    }
    __syncthreads();

    // fused LSTM gate math (PyTorch order i,f,g,o)
    for (int e = tid; e < MT * 16; e += 128) {
        int row = e >> 4, cc = e & 15;
        float gi = shG[e];
        float gf = shG[MT * 16 + e];
        float gg = shG[2 * MT * 16 + e];
        float go = shG[3 * MT * 16 + e];
        long  idx = (long)(m0 + row) * H_LSTM + (n0 + cc);
        float cold = C[idx];
        float cn = sigmoidf_fast(gf) * cold + sigmoidf_fast(gi) * tanhf(gg);
        float hv = sigmoidf_fast(go) * tanhf(cn);
        C[idx]    = cn;
        Hnew[idx] = (_Float16)hv;
    }
}

// --- MLP layer 1: relu([Hpep|Hhla] @ W1^T + b1), f16 out -------------------
__global__ __launch_bounds__(128) void k_mlp1(
    const _Float16* __restrict__ Hpep, const _Float16* __restrict__ Hhla,
    const _Float16* __restrict__ W1h,  // [1024][1024] f16
    const float*    __restrict__ b1,
    _Float16*       __restrict__ H1)   // [4096][1024] f16
{
    const int tid  = threadIdx.x;
    const int wave = tid >> 5;
    const int lane = tid & 31;
    const int col  = lane & 15;
    const int m0   = blockIdx.x * 64;
    const int n0   = blockIdx.y * 64 + wave * 16;
    const int wrow = n0 + col;

    v8f acc[4] = {};
    for (int k0 = 0; k0 < 2 * H_LSTM; k0 += 32) {
        const _Float16* Asrc = (k0 < H_LSTM) ? Hpep : Hhla;
        const int kk = k0 & (H_LSTM - 1);
        v16h bfrag = load_frag_g(W1h, 2 * H_LSTM, wrow, k0, lane);
        #pragma unroll
        for (int a = 0; a < 4; ++a) {
            v16h afrag = load_frag_g(Asrc, H_LSTM, m0 + a * 16 + col, kk, lane);
            acc[a] = wmma_f16(afrag, bfrag, acc[a]);
        }
    }
    const float bv = b1[wrow];
    #pragma unroll
    for (int a = 0; a < 4; ++a) {
        #pragma unroll
        for (int r = 0; r < 8; ++r) {
            int row = a * 16 + r + ((lane >> 4) << 3);
            float h = acc[a][r] + bv;
            h = h > 0.0f ? h : 0.0f;
            H1[(long)(m0 + row) * H_MLP + wrow] = (_Float16)h;
        }
    }
}

// --- MLP layer 2: out[b] = H1[b]·W2 + b2, one wave per row ------------------
__global__ __launch_bounds__(128) void k_mlp2(
    const _Float16* __restrict__ H1, const float* __restrict__ W2,
    const float* __restrict__ b2, float* __restrict__ out)
{
    int row  = blockIdx.x * 4 + (threadIdx.x >> 5);
    int lane = threadIdx.x & 31;
    float s = 0.0f;
    for (int k = lane; k < H_MLP; k += 32)
        s += (float)H1[(long)row * H_MLP + k] * W2[k];
    #pragma unroll
    for (int off = 16; off > 0; off >>= 1)
        s += __shfl_down(s, off, 32);
    if (lane == 0) out[row] = s + b2[0];
}

// ---------------------------------------------------------------------------
extern "C" void kernel_launch(void* const* d_in, const int* in_sizes, int n_in,
                              void* d_out, int out_size, void* d_ws, size_t ws_size,
                              hipStream_t stream) {
    const float* input_pep = (const float*)d_in[0];
    const float* input_hla = (const float*)d_in[1];
    const float* W_ih_pep  = (const float*)d_in[2];
    const float* W_hh_pep  = (const float*)d_in[3];
    const float* b_ih_pep  = (const float*)d_in[4];
    const float* b_hh_pep  = (const float*)d_in[5];
    const float* W_ih_hla  = (const float*)d_in[6];
    const float* W_hh_hla  = (const float*)d_in[7];
    const float* b_ih_hla  = (const float*)d_in[8];
    const float* b_hh_hla  = (const float*)d_in[9];
    const float* W1        = (const float*)d_in[10];
    const float* b1        = (const float*)d_in[11];
    const float* W2        = (const float*)d_in[12];
    const float* b2        = (const float*)d_in[13];
    float* out = (float*)d_out;

    char* w = (char*)d_ws;
    size_t off = 0;
    auto alloc = [&](size_t bytes) -> char* {
        char* p = w + off;
        off += (bytes + 255) & ~(size_t)255;
        return p;
    };
    _Float16* WhhP = (_Float16*)alloc((size_t)G4 * H_LSTM * 2);
    _Float16* WhhH = (_Float16*)alloc((size_t)G4 * H_LSTM * 2);
    _Float16* WihP = (_Float16*)alloc((size_t)G4 * KPAD * 2);
    _Float16* WihH = (_Float16*)alloc((size_t)G4 * KPAD * 2);
    _Float16* W1h  = (_Float16*)alloc((size_t)H_MLP * 2 * H_LSTM * 2);
    float*    bP   = (float*)   alloc((size_t)G4 * 4);
    float*    bH   = (float*)   alloc((size_t)G4 * 4);
    _Float16* XP   = (_Float16*)alloc((size_t)L_PEP * B_SZ * KPAD * 2);
    _Float16* XH   = (_Float16*)alloc((size_t)L_HLA * B_SZ * KPAD * 2);
    _Float16* HP0  = (_Float16*)alloc((size_t)B_SZ * H_LSTM * 2);
    _Float16* HP1  = (_Float16*)alloc((size_t)B_SZ * H_LSTM * 2);
    _Float16* HH0  = (_Float16*)alloc((size_t)B_SZ * H_LSTM * 2);
    _Float16* HH1  = (_Float16*)alloc((size_t)B_SZ * H_LSTM * 2);
    float*    CP   = (float*)   alloc((size_t)B_SZ * H_LSTM * 4);
    float*    CH   = (float*)   alloc((size_t)B_SZ * H_LSTM * 4);
    _Float16* H1b  = (_Float16*)alloc((size_t)B_SZ * H_MLP * 2);
    (void)ws_size; (void)in_sizes; (void)n_in; (void)out_size;

    // weight conversion / padding / bias combine
    k_cvt_f16<<<512, 256, 0, stream>>>(W_hh_pep, WhhP, G4 * H_LSTM);
    k_cvt_f16<<<512, 256, 0, stream>>>(W_hh_hla, WhhH, G4 * H_LSTM);
    k_cvt_f16<<<512, 256, 0, stream>>>(W1, W1h, H_MLP * 2 * H_LSTM);
    k_cvt_pad20<<<256, 256, 0, stream>>>(W_ih_pep, WihP, G4);
    k_cvt_pad20<<<256, 256, 0, stream>>>(W_ih_hla, WihH, G4);
    k_cvt_pad20<<<1024, 256, 0, stream>>>(input_pep, XP, L_PEP * B_SZ);
    k_cvt_pad20<<<1024, 256, 0, stream>>>(input_hla, XH, L_HLA * B_SZ);
    k_bias_comb<<<8, 256, 0, stream>>>(b_ih_pep, b_hh_pep, bP, G4);
    k_bias_comb<<<8, 256, 0, stream>>>(b_ih_hla, b_hh_hla, bH, G4);

    // zero-init h0 / c0 every call (graph replays, no re-poison)
    k_zero32<<<512, 256, 0, stream>>>((unsigned int*)HP0, B_SZ * H_LSTM / 2);
    k_zero32<<<512, 256, 0, stream>>>((unsigned int*)HH0, B_SZ * H_LSTM / 2);
    k_zero32<<<512, 256, 0, stream>>>((unsigned int*)CP, B_SZ * H_LSTM);
    k_zero32<<<512, 256, 0, stream>>>((unsigned int*)CH, B_SZ * H_LSTM);

    dim3 sgrid(B_SZ / MT, H_LSTM / 16);

    _Float16* hp[2] = {HP0, HP1};
    for (int t = 0; t < L_PEP; ++t)
        k_lstm_step<<<sgrid, 128, 0, stream>>>(
            hp[t & 1], WhhP, XP + (size_t)t * B_SZ * KPAD, WihP, bP, CP, hp[(t + 1) & 1]);
    _Float16* hpep_final = hp[L_PEP & 1];

    _Float16* hh[2] = {HH0, HH1};
    for (int t = 0; t < L_HLA; ++t)
        k_lstm_step<<<sgrid, 128, 0, stream>>>(
            hh[t & 1], WhhH, XH + (size_t)t * B_SZ * KPAD, WihH, bH, CH, hh[(t + 1) & 1]);
    _Float16* hhla_final = hh[L_HLA & 1];

    dim3 mgrid(B_SZ / 64, H_MLP / 64);
    k_mlp1<<<mgrid, 128, 0, stream>>>(hpep_final, hhla_final, W1h, b1, H1b);
    k_mlp2<<<B_SZ / 4, 128, 0, stream>>>(H1b, W2, b2, out);
}
